// FFTDiffuseBase_85478439125951
// MI455X (gfx1250) — matrix operand order repaired and checked by reading, not compiled, not run on hardware
//
#include <hip/hip_runtime.h>

// ---------------- problem constants ----------------
#define HH 512
#define WW 512
#define BATCH 2
#define CH_G 3
#define NPIX (HH*WW)              // 262144
#define N_YP (BATCH*NPIX)         // 524288 (y_pred elems)
#define HC (HH-1)                 // 511 (cv rows)
#define WCH (WW-1)                // 511 (ch cols)
#define N_CV (BATCH*HC*WW)        // 523264
#define N_CH (BATCH*HH*WCH)       // 523264
#define N_UNI (BATCH*HC*WCH)      // 522242

#define LV 0.24f
#define KSQ (0.03f*0.03f)
#define DEPS_ 0.1f
#define THRESH_ 0.1f
#define FFT_STEPS_ 10
#define NITER_ 64
#define BLK 64
#define OVL 16
#define STEP 48
#define NB 11                      // range(0,512,48) -> 11 block origins

typedef __attribute__((ext_vector_type(16))) _Float16 v16h;
typedef __attribute__((ext_vector_type(8)))  _Float16 v8h;
typedef __attribute__((ext_vector_type(8)))  float    v8f;
typedef __attribute__((ext_vector_type(4)))  unsigned int u32x4;
typedef __attribute__((ext_vector_type(8)))  int i32x8;
typedef __attribute__((ext_vector_type(4)))  int i32x4;

// Fragment-layout staging indices (wave32 WMMA; this code's contiguous k order, same for A and B).
// A-layout (16x32 f16 tile): element (m,k) -> buffer position
//   lane = (k%32 >= 16)*16 + m%16 ; half i = k%16
__device__ inline int a_idx(int m, int k) {
  const int trg = m >> 4, mm = m & 15;
  const int kb2 = k >> 5, kk = k & 31;
  const int grp = kk >> 4;
  const int i = kk & 15;
  return ((kb2 * 4 + trg) * 32 + (grp * 16 + mm)) * 16 + i;
}
// B-layout (32x16 f16 tile): element (k,n) -> lane = (k%32>=16)*16 + n%16, i = k%16
__device__ inline int b_idx(int k, int n) {
  const int tcg = n >> 4, nn = n & 15;
  const int kb2 = k >> 5, kk = k & 31;
  return ((kb2 * 4 + tcg) * 32 + ((kk >> 4) * 16 + nn)) * 16 + (kk & 15);
}

// ---------------- small helpers ----------------
__device__ inline float block_reduce_sum(float v, float* red) {
  const int tid = threadIdx.x;
  red[tid] = v;
  __syncthreads();
  for (int s = 128; s > 0; s >>= 1) {
    if (tid < s) red[tid] += red[tid + s];
    __syncthreads();
  }
  float r = red[0];
  __syncthreads();
  return r;
}

__device__ inline float gfun(float x) { return 1.0f / (1.0f + (x * x) / KSQ); }

// ---------------- stage 0: shift = DEPS if min(initial) <= DEPS else 0 -----
__global__ void ws_init_kernel(unsigned* minbits) {
  if (blockIdx.x == 0 && threadIdx.x == 0) minbits[0] = 0x7f7fffffu; // +FLT_MAX
}

__global__ void min_reduce_kernel(const float* __restrict__ ini, unsigned* minbits) {
  __shared__ unsigned red[256];
  const int tid = threadIdx.x;
  unsigned best = 0x7f7fffffu;
  for (int i = blockIdx.x * blockDim.x + tid; i < N_YP; i += gridDim.x * blockDim.x) {
    unsigned u = __float_as_uint(ini[i]); // inputs uniform[0,1): nonneg -> bit order == float order
    if (u < best) best = u;
  }
  red[tid] = best;
  __syncthreads();
  for (int s = 128; s > 0; s >>= 1) {
    if (tid < s) { if (red[tid + s] < red[tid]) red[tid] = red[tid + s]; }
    __syncthreads();
  }
  if (tid == 0) atomicMin(minbits, red[0]);
}

__global__ void shift_kernel(const unsigned* minbits, float* shift) {
  if (blockIdx.x == 0 && threadIdx.x == 0) {
    float mn = __uint_as_float(minbits[0]);
    shift[0] = (mn <= DEPS_) ? DEPS_ : 0.0f;
  }
}

// ---------------- stage 1: cv / ch (written straight into d_out) -----------
__global__ void cvch_kernel(const float* __restrict__ guide, float* __restrict__ cv,
                            float* __restrict__ ch) {
  const int idx = blockIdx.x * blockDim.x + threadIdx.x;
  if (idx >= N_YP) return;
  const int b = idx / NPIX, r = idx % NPIX;
  const int i = r / WW, j = r % WW;
  const float* g = guide + (size_t)b * CH_G * NPIX;
  if (i < HH - 1) {
    float s = 0.f;
#pragma unroll
    for (int c = 0; c < CH_G; ++c)
      s += fabsf(g[c * NPIX + (i + 1) * WW + j] - g[c * NPIX + i * WW + j]);
    cv[b * HC * WW + i * WW + j] = gfun(s * (1.0f / 3.0f));
  }
  if (j < WW - 1) {
    float s = 0.f;
#pragma unroll
    for (int c = 0; c < CH_G; ++c)
      s += fabsf(g[c * NPIX + i * WW + j + 1] - g[c * NPIX + i * WW + j]);
    ch[b * HH * WCH + i * WCH + j] = gfun(s * (1.0f / 3.0f));
  }
}

// ---------------- stage 2: uniform mask (zero-padded 3x3 variance) ---------
__global__ void uniform_kernel(const float* __restrict__ cv, const float* __restrict__ ch,
                               float* __restrict__ uni) {
  const int idx = blockIdx.x * blockDim.x + threadIdx.x;
  if (idx >= N_UNI) return;
  const int b = idx / (HC * WCH), r = idx % (HC * WCH);
  const int i = r / WCH, j = r % WCH;

  float s = 0.f, s2 = 0.f;
#pragma unroll
  for (int di = -1; di <= 1; ++di)
#pragma unroll
    for (int dj = -1; dj <= 1; ++dj) {
      const int ii = i + di, jj = j + dj;
      if (ii >= 0 && ii < HC && jj >= 0 && jj < WW) {
        float v = cv[b * HC * WW + ii * WW + jj];
        s += v; s2 += v * v;
      }
    }
  const float cvm = s * (1.0f / 9.0f);
  const float cvvar = s2 * (1.0f / 9.0f) - cvm * cvm;

  s = 0.f; s2 = 0.f;
#pragma unroll
  for (int di = -1; di <= 1; ++di)
#pragma unroll
    for (int dj = -1; dj <= 1; ++dj) {
      const int ii = i + di, jj = j + dj;
      if (ii >= 0 && ii < HH && jj >= 0 && jj < WCH) {
        float v = ch[b * HH * WCH + ii * WCH + jj];
        s += v; s2 += v * v;
      }
    }
  const float chm = s * (1.0f / 9.0f);
  const float chvar = s2 * (1.0f / 9.0f) - chm * chm;

  uni[idx] = (cvvar < THRESH_ && chvar < THRESH_) ? 1.0f : 0.0f;
}

// ---------------- stage 3: depth = initial + shift -------------------------
__global__ void depth_init_kernel(const float* __restrict__ ini, const float* shift,
                                  float* __restrict__ depth) {
  const int idx = blockIdx.x * blockDim.x + threadIdx.x;
  if (idx < N_YP) depth[idx] = ini[idx] + shift[0];
}

// ---------------- stage 4: sequential overlapped block solve ---------------
// irfft2(rfft2(F)*op) == M^10 F with M F = F - KY*F - F*KX (symmetric
// circulant stencils; 1-cos(k) is the DFT eigenvalue of I - 0.5(S+S^-1)).
// 10 x (two 64x64x64 GEMMs) via v_wmma_f32_16x16x32_f16, operands staged in
// LDS pre-swizzled into the wave32 WMMA register layouts so every fragment is
// a single aligned 32B LDS read (2x ds_load_b128). Each wave keeps its two
// 16x16 D-tiles (== its F elements) in f32 registers across all 10 steps.
// Block tile is DMA'd into LDS by the Tensor Data Mover.
__global__ __launch_bounds__(256) void block_fft_kernel(float* __restrict__ depth,
                                                        const float* __restrict__ cv,
                                                        const float* __restrict__ ch,
                                                        const float* __restrict__ uni) {
  __shared__ __align__(32) float    sO[BLK][BLK];       // TDM landing + original block
  __shared__ __align__(32) _Float16 sAK[2 * 4 * 32 * 16]; // KY, A-layout (static per block)
  __shared__ __align__(32) _Float16 sBX[2 * 4 * 32 * 16]; // KX, B-layout (static per block)
  __shared__ __align__(32) _Float16 sAF[2 * 4 * 32 * 16]; // F,  A-layout (per step)
  __shared__ __align__(32) _Float16 sBF[2 * 4 * 32 * 16]; // F,  B-layout (per step)
  __shared__ float red[256];

  const int b    = blockIdx.x;
  const int tid  = threadIdx.x;
  const int lane = tid & 31;
  const int wave = tid >> 5;

  const float* cvb = cv  + (size_t)b * HC * WW;
  const float* chb = ch  + (size_t)b * HH * WCH;
  const float* unb = uni + (size_t)b * HC * WCH;
  float*       dpb = depth + (size_t)b * NPIX;

  for (int byi = 0; byi < NB; ++byi) {
    const int y  = byi * STEP;
    const int ye = (y + BLK < HH) ? (y + BLK) : HH;
    const int bh = ye - y;
    for (int bxi = 0; bxi < NB; ++bxi) {
      const int x  = bxi * STEP;
      const int xe = (x + BLK < WW) ? (x + BLK) : WW;
      const int bw = xe - x;

      __syncthreads(); // separate from previous block's writeback

      // ---- zero the landing pad (TDM pad-skips + short tiles leave gaps) --
      for (int idx = tid; idx < BLK * BLK; idx += 256) ((float*)sO)[idx] = 0.0f;
      __syncthreads();

      // ---- Tensor Data Mover: bh x bw f32 tile, row stride 512, into sO ---
      if (wave == 0) {
        const unsigned lds_addr = (unsigned)(unsigned long long)(&sO[0][0]);
        const unsigned long long ga =
            (unsigned long long)(dpb + (size_t)y * WW + x);
        u32x4 g0;
        g0[0] = 1u;                                        // count=1 (valid user D#)
        g0[1] = lds_addr;                                  // lds_addr [63:32]
        g0[2] = (unsigned)(ga & 0xffffffffu);              // global_addr lo
        g0[3] = (unsigned)((ga >> 32) & 0x01ffffffu)       // global_addr [56:32]
              | (2u << 30);                                // type=2 ("image")
        unsigned w0 = (2u << 16);                          // data_size = 4B
        if (bw == 32)                                      // repitch 32-wide rows to 64 floats
          w0 |= (1u << 20)            /* pad_enable   */
              | (4u << 22)            /* interval = 32 DWORDs */
              | (31u << 25);          /* amount   = 32 DWORDs */
        i32x8 g1;
        g1[0] = (int)w0;                                   // wg_mask=0 (not in cluster)
        g1[1] = (int)((unsigned)(WW & 0xffff) << 16);      // tensor_dim0 lo16 @ [63:48]
        g1[2] = (int)(((unsigned)WW >> 16) | ((unsigned)(HH & 0xffff) << 16)); // dim0 hi | dim1 lo
        g1[3] = (int)(((unsigned)HH >> 16) | ((unsigned)bw << 16)); // dim1 hi | tile_dim0
        g1[4] = (int)(unsigned)bh;                         // tile_dim1 (tile_dim2 = 0)
        g1[5] = (int)WW;                                   // tensor_dim0_stride lo32
        g1[6] = 0;
        g1[7] = 0;
        const i32x4 gz = (i32x4)0;                         // <=2D tensor: groups 2/3 unused
#if __has_include(<hip/amd_detail/amd_gfx1250_TDM.h>)      // therock clang-23: 6-arg form
        const i32x8 gz8 = (i32x8)0;
        __builtin_amdgcn_tensor_load_to_lds(g0, g1, gz, gz, gz8, 0);
#else                                                      // ROCm 7.2 clang-22: 5-arg form
        __builtin_amdgcn_tensor_load_to_lds(g0, g1, gz, gz, 0);
#endif
        __builtin_amdgcn_s_wait_tensorcnt(0);
      }
      __syncthreads();

      // ---- block means: cv, ch, uniform ----
      float scv = 0.f;
      for (int idx = tid; idx < (bh - 1) * bw; idx += 256) {
        const int m = idx / bw, n = idx % bw;
        scv += cvb[(y + m) * WW + (x + n)];
      }
      scv = block_reduce_sum(scv, red);

      float sch = 0.f;
      for (int idx = tid; idx < bh * (bw - 1); idx += 256) {
        const int m = idx / (bw - 1), n = idx % (bw - 1);
        sch += chb[(y + m) * WCH + (x + n)];
      }
      sch = block_reduce_sum(sch, red);

      const int uh = ((ye < HC) ? ye : HC) - y;
      const int uw = ((xe < WCH) ? xe : WCH) - x;
      float sun = 0.f;
      for (int idx = tid; idx < uh * uw; idx += 256) {
        const int m = idx / uw, n = idx % uw;
        sun += unb[(y + m) * WCH + (x + n)];
      }
      sun = block_reduce_sum(sun, red);

      const bool cond = (sun / (float)(uh * uw)) > 0.7f; // workgroup-uniform
      if (!cond) continue;                               // depth unchanged

      const float ay = 2.0f * LV * (scv / (float)((bh - 1) * bw));
      const float ax = 2.0f * LV * (sch / (float)(bh * (bw - 1)));

      // ---- stage KY (A-layout), KX (B-layout), F (both layouts) -----------
      for (int idx = tid; idx < BLK * BLK; idx += 256) {
        const int p = idx >> 6, q = idx & 63;
        float vy = 0.f, vx = 0.f;
        if (p < bh && q < bh) {
          if (q == p) vy = ay;
          else if (q == (p + 1) % bh || q == (p - 1 + bh) % bh) vy = -0.5f * ay;
        }
        if (p < bw && q < bw) {
          if (q == p) vx = ax;
          else if (q == (p + 1) % bw || q == (p - 1 + bw) % bw) vx = -0.5f * ax;
        }
        sAK[a_idx(p, q)] = (_Float16)vy;  // KY element (m=p, k=q)
        sBX[b_idx(p, q)] = (_Float16)vx;  // KX element (k=p, n=q)
        const _Float16 h = (_Float16)sO[p][q];
        sAF[a_idx(p, q)] = h;
        sBF[b_idx(p, q)] = h;
      }

      // ---- each wave owns two 16x16 F tiles in registers (C/D layout) -----
      float Freg[2][8];
      {
        const int mo = (lane & 16) ? 8 : 0;
#pragma unroll
        for (int t = 0; t < 2; ++t) {
          const int tile = wave * 2 + t;
          const int tr = (tile >> 2) * 16, tc = (tile & 3) * 16;
          const int n = tc + (lane & 15);
#pragma unroll
          for (int r = 0; r < 8; ++r) Freg[t][r] = sO[tr + mo + r][n];
        }
      }
      __syncthreads();

      // ---- F <- F - KY*F - F*KX, ten times, via WMMA ----------------------
      for (int s = 0; s < FFT_STEPS_; ++s) {
        v8f a1[2], a2[2];
#pragma unroll
        for (int t = 0; t < 2; ++t) {
          const int tile = wave * 2 + t;
          const int trg = tile >> 2, tcg = tile & 3;
          v8f acc1 = {}; v8f acc2 = {};
#pragma unroll
          for (int kb2 = 0; kb2 < 2; ++kb2) {
            const v16h aK = *(const v16h*)&sAK[((kb2 * 4 + trg) * 32 + lane) * 16];
            const v16h bF = *(const v16h*)&sBF[((kb2 * 4 + tcg) * 32 + lane) * 16];
            acc1 = __builtin_amdgcn_wmma_f32_16x16x32_f16(false, aK, false, bF,
                                                          (short)0, acc1, false, false);
            const v16h aF = *(const v16h*)&sAF[((kb2 * 4 + trg) * 32 + lane) * 16];
            const v16h bX = *(const v16h*)&sBX[((kb2 * 4 + tcg) * 32 + lane) * 16];
            acc2 = __builtin_amdgcn_wmma_f32_16x16x32_f16(false, aF, false, bX,
                                                          (short)0, acc2, false, false);
          }
          a1[t] = acc1; a2[t] = acc2;
        }
        __syncthreads(); // all fragment reads done before restaging
#pragma unroll
        for (int t = 0; t < 2; ++t) {
          const int tile = wave * 2 + t;
          const int tr = (tile >> 2) * 16, tcg = tile & 3;
          const int mo = (lane & 16) ? 8 : 0;
          const int n  = tcg * 16 + (lane & 15);
          v8h pack;
#pragma unroll
          for (int r = 0; r < 8; ++r) {
            const int m = tr + mo + r;
            const float nv = Freg[t][r] - a1[t][r] - a2[t][r];
            Freg[t][r] = nv;
            const _Float16 h = (_Float16)nv;
            sAF[a_idx(m, n)] = h;   // scattered 16-bit stores (stride 32B)
            pack[r] = h;            // B-layout run is contiguous: i = mo..mo+7
          }
          const int kb2  = tr >> 5;
          const int lgrp = (tr >> 4) & 1;
          *(v8h*)&sBF[(((kb2 * 4 + tcg) * 32) + (lgrp * 16 + (lane & 15))) * 16 + mo] = pack;
        }
        __syncthreads();
      }

      // ---- overlap blend + writeback from registers -----------------------
      {
        const int mo = (lane & 16) ? 8 : 0;
#pragma unroll
        for (int t = 0; t < 2; ++t) {
          const int tile = wave * 2 + t;
          const int tr = (tile >> 2) * 16, tc = (tile & 3) * 16;
          const int n = tc + (lane & 15);
#pragma unroll
          for (int r = 0; r < 8; ++r) {
            const int m = tr + mo + r;
            if (m < bh && n < bw) {
              const float byf = (y > 0 && m < OVL) ? (float)m * (1.0f / OVL) : 1.0f;
              const float bxf = (x > 0 && n < OVL) ? (float)n * (1.0f / OVL) : 1.0f;
              const float bl = byf * bxf;
              dpb[(y + m) * WW + (x + n)] = sO[m][n] * (1.0f - bl) + Freg[t][r] * bl;
            }
          }
        }
      }
      __threadfence(); // publish before next block's TDM read
    }
  }
}

// ---------------- stage 5: fused explicit diffusion iteration --------------
__global__ void diffuse_kernel(const float* __restrict__ I, float* __restrict__ O,
                               const float* __restrict__ cv, const float* __restrict__ ch) {
  const int idx = blockIdx.x * blockDim.x + threadIdx.x;
  if (idx >= N_YP) return;
  const int b = idx / NPIX, r = idx % NPIX;
  const int i = r / WW, j = r % WW;
  const float* Ib  = I  + (size_t)b * NPIX;
  const float* cvb = cv + (size_t)b * HC * WW;
  const float* chb = ch + (size_t)b * HH * WCH;

  __builtin_prefetch(&Ib[r], 0, 1); // global_prefetch

  auto tv = [&](int ii, int jj) -> float {
    if (ii < 0 || ii >= HH - 1) return 0.0f;
    return LV * cvb[ii * WW + jj] * (Ib[(ii + 1) * WW + jj] - Ib[ii * WW + jj]);
  };
  auto U = [&](int jj) -> float { // vertically-updated field at (i, jj)
    return Ib[i * WW + jj] + tv(i, jj) - tv(i - 1, jj);
  };
  auto th = [&](int jj) -> float {
    if (jj < 0 || jj >= WW - 1) return 0.0f;
    return LV * chb[i * WCH + jj] * (U(jj + 1) - U(jj));
  };
  O[idx] = U(j) + th(j) - th(j - 1);
}

// ---------------- stage 6: y_pred = depth - shift --------------------------
__global__ void finalize_kernel(const float* __restrict__ depth, const float* shift,
                                float* __restrict__ ypred) {
  const int idx = blockIdx.x * blockDim.x + threadIdx.x;
  if (idx < N_YP) ypred[idx] = depth[idx] - shift[0];
}

// ---------------- host orchestration ---------------------------------------
extern "C" void kernel_launch(void* const* d_in, const int* in_sizes, int n_in,
                              void* d_out, int out_size, void* d_ws, size_t ws_size,
                              hipStream_t stream) {
  (void)in_sizes; (void)n_in; (void)out_size; (void)ws_size;
  const float* guide   = (const float*)d_in[0];
  const float* initial = (const float*)d_in[1];

  float* out   = (float*)d_out;
  float* ypred = out;                 // (2,1,512,512)
  float* cv    = out + N_YP;          // (2,1,511,512)
  float* ch    = cv + N_CV;           // (2,1,512,511)

  float*    wsf     = (float*)d_ws;
  unsigned* minbits = (unsigned*)d_ws;     // wsf[0]
  float*    shift   = wsf + 1;             // wsf[1]
  float*    depthA  = wsf + 64;
  float*    depthB  = depthA + N_YP;
  float*    uni     = depthB + N_YP;       // (2,1,511,511)

  const int T = 256;
  const int gN = (N_YP + T - 1) / T;
  const int gU = (N_UNI + T - 1) / T;

  ws_init_kernel<<<1, 1, 0, stream>>>(minbits);
  min_reduce_kernel<<<512, T, 0, stream>>>(initial, minbits);
  shift_kernel<<<1, 1, 0, stream>>>(minbits, shift);
  cvch_kernel<<<gN, T, 0, stream>>>(guide, cv, ch);
  uniform_kernel<<<gU, T, 0, stream>>>(cv, ch, uni);
  depth_init_kernel<<<gN, T, 0, stream>>>(initial, shift, depthA);
  block_fft_kernel<<<BATCH, T, 0, stream>>>(depthA, cv, ch, uni);

  float* a = depthA; float* bb = depthB;
  for (int it = 0; it < NITER_; ++it) {
    diffuse_kernel<<<gN, T, 0, stream>>>(a, bb, cv, ch);
    float* t = a; a = bb; bb = t;
  }
  finalize_kernel<<<gN, T, 0, stream>>>(a, shift, ypred);
}
